// CausalSparseAttention_52956946760511
// MI455X (gfx1250) — compile-verified, matrix operand order checked by reading, not run on hardware
//
#include <hip/hip_runtime.h>
#include <hip/hip_bf16.h>
#include <stdint.h>

// ---------------------------------------------------------------------------
// CausalSparseAttention decode step, restructured for MI455X (gfx1250):
//  - one streaming pass over k_cache computing per-head raw scores via WMMA
//  - TDM (tensor_load_to_lds) stages 16x1024 f32 tiles into LDS
//  - exact per-head top-12288 (radix select + LDS bitonic sort, jax tie order)
//  - chunk scores & final attention reconstructed algebraically from s[h][tok]
//    so k_new/v_new are never materialized (saves ~270MB of HBM traffic)
// ---------------------------------------------------------------------------

#define CDIM   1024
#define NH     16
#define HS     64
#define CT     65536
#define WND    4096
#define PAST   (CT - WND)        // 61440
#define KEEP   12288             // MIN_KV - WINDOW
#define TOPKC  32
#define CHUNK  64
#define NCH    256               // 16384 / 64
#define KLEN   (TOPKC * CHUNK + 1) // 2049
#define SPITCH 65536

typedef float     v8f  __attribute__((ext_vector_type(8)));
typedef _Float16  v16h __attribute__((ext_vector_type(16)));
typedef _Float16  v8h  __attribute__((ext_vector_type(8)));
typedef unsigned  v4u  __attribute__((ext_vector_type(4)));
typedef unsigned  v8u  __attribute__((ext_vector_type(8)));
typedef int       v4i  __attribute__((ext_vector_type(4)));
typedef int       v8i  __attribute__((ext_vector_type(8)));

union HB { v16h h; v8u u; };

__device__ __forceinline__ unsigned ordu(float f) {
  unsigned u = __float_as_uint(f);
  return (u & 0x80000000u) ? ~u : (u | 0x80000000u);   // order-preserving map
}

// ---------------------------------------------------------------- GEMV 1024x1024
__global__ __launch_bounds__(256) void gemv1024(const float* __restrict__ W,
                                                const float* __restrict__ xv,
                                                float* __restrict__ out) {
  __shared__ __align__(16) float xs[CDIM];
  for (int i = threadIdx.x; i < CDIM; i += 256) xs[i] = xv[i];
  __syncthreads();
  const int wave = threadIdx.x >> 5, lane = threadIdx.x & 31;
  const int row  = blockIdx.x * 8 + wave;
  const float4* w4 = (const float4*)(W + (size_t)row * CDIM);
  const float4* x4 = (const float4*)xs;
  float acc = 0.f;
  for (int j = lane; j < CDIM / 4; j += 32) {
    float4 w = w4[j], x = x4[j];
    acc += w.x * x.x + w.y * x.y + w.z * x.z + w.w * x.w;
  }
#pragma unroll
  for (int off = 16; off; off >>= 1) acc += __shfl_xor(acc, off, 32);
  if (lane == 0) out[row] = acc;
}

// ------------------------------------------------- WMMA scorer over all tokens
// s[h][tok] = dot(q_h, k_cache[tok] segment h) / 8, for all 65536 tokens.
// One wave owns a 16-token tile; TDM stages the 16x1024 f32 tile to LDS.
__global__ __launch_bounds__(128) void score_wmma(const float* __restrict__ kc,
                                                  const float* __restrict__ qv,
                                                  float* __restrict__ s) {
  extern __shared__ char sm[];                 // 4 x 64KB tiles + 2KB f16 q
  const int wave = threadIdx.x >> 5, lane = threadIdx.x & 31;
  _Float16* q16 = (_Float16*)(sm + 4 * 65536);
  for (int i = threadIdx.x; i < CDIM; i += 128) q16[i] = (_Float16)qv[i];
  __syncthreads();

  const int tile = blockIdx.x * 4 + wave;      // < 4096
  char* tbuf = sm + wave * 65536;
  const float* tf = (const float*)tbuf;

  // ---- kick off the TDM transfer first, overlap descriptor + B preload ----
#if __has_builtin(__builtin_amdgcn_tensor_load_to_lds)
  {
    unsigned long long ga = (unsigned long long)(uintptr_t)kc +
                            (unsigned long long)tile * (16ull * CDIM * 4ull);
    // low 32 bits of a flat shared address are the LDS byte offset
    unsigned ldsoff = (unsigned)(uintptr_t)tbuf;
    unsigned galo = (unsigned)(ga & 0xffffffffull);
    unsigned gahi = (unsigned)((ga >> 32) & 0x01ffffffull) | 0x80000000u; // type=2
    ldsoff = __builtin_amdgcn_readfirstlane(ldsoff);
    galo   = __builtin_amdgcn_readfirstlane(galo);
    gahi   = __builtin_amdgcn_readfirstlane(gahi);
    v4u g0; g0.x = 1u;            // count=1, gather off
    g0.y = ldsoff; g0.z = galo; g0.w = gahi;
    v8i g1;
    g1[0] = (int)(2u << 16);              // data_size = 4B
    g1[1] = (int)(1024u << 16);           // tensor_dim0 = 1024
    g1[2] = 0;                            // tensor_dim0 hi / tensor_dim1 lo
    g1[3] = (int)(1u | (1024u << 16));    // tensor_dim1 = 65536, tile_dim0 = 1024
    g1[4] = 16;                           // tile_dim1 = 16, tile_dim2 = 0
    g1[5] = 1024;                         // tensor_dim0_stride = 1024
    g1[6] = 0; g1[7] = 0;
    v4i z4 = {0, 0, 0, 0};
#if defined(__clang_major__) && __clang_major__ >= 23
    v8i z8 = {0, 0, 0, 0, 0, 0, 0, 0};
    __builtin_amdgcn_tensor_load_to_lds(g0, g1, z4, z4, z8, 0);
#else
    __builtin_amdgcn_tensor_load_to_lds(g0, g1, z4, z4, 0);
#endif
  }
#else
  { // fallback: wave-cooperative copy
    const float* src = kc + (size_t)tile * 16 * CDIM;
    float* dst = (float*)tbuf;
    for (int i = lane; i < 16 * CDIM; i += 32) dst[i] = src[i];
  }
#endif

  // ---- per-lane constants -------------------------------------------------
  const int rowM = lane & 15;
  const int hi   = lane >> 4;        // 0: K 0-7,16-23 ; 1: K 8-15,24-31
  const int ncol = lane & 15;        // D column = head

  // Each lane's B column (head ncol) is non-zero only for j = 2*ncol, 2*ncol+1.
  // Preload both fragments once, unconditionally (aligned b128 LDS loads),
  // while the TDM transfer is still in flight.
  v16h bE, bO;
  {
    const _Float16* brE = q16 + (ncol << 6) + hi * 8;        // f = 64*ncol
    const _Float16* brO = brE + 32;                          // f = 64*ncol+32
    v8h e0 = *(const v8h*)(brE);
    v8h e1 = *(const v8h*)(brE + 16);
    v8h o0 = *(const v8h*)(brO);
    v8h o1 = *(const v8h*)(brO + 16);
    bE = __builtin_shufflevector(e0, e1, 0,1,2,3,4,5,6,7,8,9,10,11,12,13,14,15);
    bO = __builtin_shufflevector(o0, o1, 0,1,2,3,4,5,6,7,8,9,10,11,12,13,14,15);
  }
  HB hbE, hbO, hbZ;
  hbE.h = bE; hbO.h = bO;
  hbZ.u = (v8u){0u,0u,0u,0u,0u,0u,0u,0u};

#if __has_builtin(__builtin_amdgcn_tensor_load_to_lds)
#if __has_builtin(__builtin_amdgcn_s_wait_tensorcnt)
  __builtin_amdgcn_s_wait_tensorcnt(0);
#else
  asm volatile("s_wait_tensorcnt 0x0" ::: "memory");
#endif
  asm volatile("" ::: "memory");
#endif

  // ---- 32 x v_wmma_f32_16x16x32_f16, branchless B selection ---------------
  v8f acc = {0.f, 0.f, 0.f, 0.f, 0.f, 0.f, 0.f, 0.f};
#pragma unroll
  for (int j = 0; j < 32; ++j) {
    const int f = j << 5;
    // A: 16 tokens x 32 features (f32 tile in LDS -> f16 fragment)
    const float4* a4 = (const float4*)(tf + rowM * CDIM + f + hi * 8);
    float4 x0 = a4[0], x1 = a4[1];     // floats 0..7  (K 0-7 / 8-15)
    float4 x2 = a4[4], x3 = a4[5];     // floats 16..23 (K 16-23 / 24-31)
    v16h a;
    a[0]  = (_Float16)x0.x; a[1]  = (_Float16)x0.y;
    a[2]  = (_Float16)x0.z; a[3]  = (_Float16)x0.w;
    a[4]  = (_Float16)x1.x; a[5]  = (_Float16)x1.y;
    a[6]  = (_Float16)x1.z; a[7]  = (_Float16)x1.w;
    a[8]  = (_Float16)x2.x; a[9]  = (_Float16)x2.y;
    a[10] = (_Float16)x2.z; a[11] = (_Float16)x2.w;
    a[12] = (_Float16)x3.x; a[13] = (_Float16)x3.y;
    a[14] = (_Float16)x3.z; a[15] = (_Float16)x3.w;
    // B: j&1 is uniform (unrolled), only per-lane select is vs zero
    const v8u bsel = (j & 1) ? hbO.u : hbE.u;
    const bool active = (ncol == (j >> 1));
    HB hb; hb.u = active ? bsel : hbZ.u;       // 8x v_cndmask_b32
    acc = __builtin_amdgcn_wmma_f32_16x16x32_f16(false, a, false, hb.h,
                                                 (short)0, acc, false, false);
  }
  // D layout: lanes 0-15 -> N=lane, M=r ; lanes 16-31 -> N=lane-16, M=8+r
  const int tok0 = tile * 16 + hi * 8;
#pragma unroll
  for (int r = 0; r < 8; ++r)
    s[ncol * SPITCH + tok0 + r] = acc[r] * 0.125f;   // 1/sqrt(HS)
}

// -------------------------------- per-head exact top-KEEP, sorted like jax top_k
__global__ __launch_bounds__(1024) void topk_head(const float* __restrict__ s,
                                                  int* __restrict__ idxbuf) {
  extern __shared__ char sm[];
  unsigned long long* keys = (unsigned long long*)sm;   // 16384 * 8B
  unsigned* hist = (unsigned*)(sm + 16384 * 8);         // 256
  unsigned* ctr  = hist + 256;
  const int h = blockIdx.x, tid = threadIdx.x;
  const float* srow = s + h * SPITCH;

  // 4-round radix select: exact value of the KEEP-th largest key
  unsigned prefix = 0, kRem = KEEP;
  for (int round = 0; round < 4; ++round) {
    const int shift = 24 - 8 * round;
    const unsigned maskAbove = round ? (0xffffffffu << (shift + 8)) : 0u;
    for (int i = tid; i < 256; i += 1024) hist[i] = 0;
    __syncthreads();
    for (int i = tid; i < PAST; i += 1024) {
      unsigned u = ordu(srow[i]);
      if ((u & maskAbove) == prefix) atomicAdd(&hist[(u >> shift) & 0xffu], 1u);
    }
    __syncthreads();
    if (tid == 0) {
      unsigned cum = 0; int b = 255;
      for (; b > 0; --b) { if (cum + hist[b] >= kRem) break; cum += hist[b]; }
      ctr[0] = (unsigned)b; ctr[1] = cum;
    }
    __syncthreads();
    const unsigned b = ctr[0], cum = ctr[1];
    prefix |= (b << shift);
    kRem -= cum;
    __syncthreads();
  }
  const unsigned T = prefix, needEq = kRem;

  // compact: strictly-greater, then equals up to needEq
  if (tid == 0) { ctr[0] = 0; ctr[1] = 0; }
  __syncthreads();
  for (int i = tid; i < PAST; i += 1024) {
    unsigned u = ordu(srow[i]);
    if (u > T) {
      unsigned p = atomicAdd(&ctr[0], 1u);
      keys[p] = ((unsigned long long)u << 32) | (unsigned)(~i);
    }
  }
  __syncthreads();
  const unsigned gt = ctr[0];
  for (int i = tid; i < PAST; i += 1024) {
    unsigned u = ordu(srow[i]);
    if (u == T) {
      unsigned p = atomicAdd(&ctr[1], 1u);
      if (p < needEq)
        keys[gt + p] = ((unsigned long long)u << 32) | (unsigned)(~i);
    }
  }
  __syncthreads();
  for (int i = tid; i < 16384; i += 1024) if (i >= KEEP) keys[i] = 0ull;
  __syncthreads();

  // bitonic sort, descending; low word ~idx -> ties resolve to smallest index
  for (int kk = 2; kk <= 16384; kk <<= 1)
    for (int j = kk >> 1; j; j >>= 1) {
      for (int t = tid; t < 16384; t += 1024) {
        int p = t ^ j;
        if (p > t) {
          bool asc = (t & kk) != 0;   // flipped -> overall descending
          unsigned long long A = keys[t], B = keys[p];
          if ((A > B) == asc) { keys[t] = B; keys[p] = A; }
        }
      }
      __syncthreads();
    }

  for (int t = tid; t < KEEP; t += 1024)
    idxbuf[h * KEEP + t] = (int)(~(unsigned)keys[t]);
}

// -------------------- chunk scores: cs[c] = (1/8) * sum_{h,t} s[h][token(h,c,t)]
__global__ __launch_bounds__(256) void chunk_scores(const float* __restrict__ s,
                                                    const int* __restrict__ idxbuf,
                                                    float* __restrict__ cs) {
  __shared__ float red[256];
  const int c = blockIdx.x;
  float a = 0.f;
  for (int i = threadIdx.x; i < NH * CHUNK; i += 256) {
    int h = i >> 6, t = i & 63;
    int pos = c * CHUNK + t;
    int tok = (c < KEEP / CHUNK) ? idxbuf[h * KEEP + pos] : (PAST + pos - KEEP);
    a += s[h * SPITCH + tok];
  }
  red[threadIdx.x] = a; __syncthreads();
  for (int o = 128; o; o >>= 1) {
    if (threadIdx.x < o) red[threadIdx.x] += red[threadIdx.x + o];
    __syncthreads();
  }
  if (threadIdx.x == 0) cs[c] = red[0] * 0.125f;
}

// ----------------------------------------------- top-32 chunks (sorted, ties->low c)
__global__ __launch_bounds__(256) void topk_chunks(const float* __restrict__ cs,
                                                   int* __restrict__ tidx) {
  __shared__ unsigned long long k[256];
  const int t = threadIdx.x;
  k[t] = ((unsigned long long)ordu(cs[t]) << 32) | (unsigned)(~t);
  __syncthreads();
  for (int kk = 2; kk <= 256; kk <<= 1)
    for (int j = kk >> 1; j; j >>= 1) {
      int p = t ^ j;
      if (p > t) {
        bool asc = (t & kk) != 0;
        unsigned long long A = k[t], B = k[p];
        if ((A > B) == asc) { k[t] = B; k[p] = A; }
      }
      __syncthreads();
    }
  if (t < TOPKC) tidx[t] = (int)(~(unsigned)k[t]);
}

// --------------------------- final attention: softmax over 2049 keys, gather V rows
__global__ __launch_bounds__(256) void final_attn(const float* __restrict__ s,
                                                  const int* __restrict__ idxbuf,
                                                  const int* __restrict__ tidx,
                                                  const float* __restrict__ qv,
                                                  const float* __restrict__ kv,
                                                  const float* __restrict__ vv,
                                                  const float* __restrict__ v_cache,
                                                  float* __restrict__ yh) {
  __shared__ float lg[KLEN];
  __shared__ int   tk[KLEN - 1];
  __shared__ float red[256];
  __shared__ float part[256];
  const int h = blockIdx.x, tid = threadIdx.x;

  // new-token logit: dot(q_h, k_h)/8
  float a0 = 0.f;
  if (tid < HS) a0 = qv[h * HS + tid] * kv[h * HS + tid];
  red[tid] = a0; __syncthreads();
  for (int o = 128; o; o >>= 1) { if (tid < o) red[tid] += red[tid + o]; __syncthreads(); }
  if (tid == 0) lg[KLEN - 1] = red[0] * 0.125f;

  for (int e = tid; e < KLEN - 1; e += 256) {
    int ci  = tidx[e >> 6];
    int pos = ci * CHUNK + (e & 63);
    int tok = (pos < KEEP) ? idxbuf[h * KEEP + pos] : (PAST + pos - KEEP);
    tk[e] = tok;
    lg[e] = s[h * SPITCH + tok];          // already /sqrt(HS)
  }
  __syncthreads();

  float m = -3.0e38f;
  for (int e = tid; e < KLEN; e += 256) m = fmaxf(m, lg[e]);
  red[tid] = m; __syncthreads();
  for (int o = 128; o; o >>= 1) { if (tid < o) red[tid] = fmaxf(red[tid], red[tid + o]); __syncthreads(); }
  m = red[0]; __syncthreads();

  float zs = 0.f;
  for (int e = tid; e < KLEN; e += 256) { float p = __expf(lg[e] - m); lg[e] = p; zs += p; }
  red[tid] = zs; __syncthreads();
  for (int o = 128; o; o >>= 1) { if (tid < o) red[tid] += red[tid + o]; __syncthreads(); }
  const float Z = red[0];
  __syncthreads();

  const int d = tid & 63, g = tid >> 6;
  float acc = 0.f;
  for (int e = g; e < KLEN - 1; e += 4) {
    const float* vrow = v_cache + (size_t)tk[e] * CDIM + h * HS;
    __builtin_prefetch(vrow + 4 * CDIM, 0, 0);   // global_prefetch_b8
    acc += lg[e] * vrow[d];
  }
  if (g == 0) acc += lg[KLEN - 1] * vv[h * HS + d];
  part[tid] = acc; __syncthreads();
  if (tid < 64)
    yh[h * HS + tid] = (part[tid] + part[tid + 64] + part[tid + 128] + part[tid + 192]) / Z;
}

// ---------------------------------------------------------------------------
extern "C" void kernel_launch(void* const* d_in, const int* in_sizes, int n_in,
                              void* d_out, int out_size, void* d_ws, size_t ws_size,
                              hipStream_t stream) {
  (void)in_sizes; (void)n_in; (void)out_size; (void)ws_size;
  const float* x  = (const float*)d_in[0];
  const float* kc = (const float*)d_in[1];
  const float* vc = (const float*)d_in[2];
  const float* Wr = (const float*)d_in[3];
  const float* Wk = (const float*)d_in[4];
  const float* Wv = (const float*)d_in[5];
  const float* Wo = (const float*)d_in[6];

  float* ws   = (float*)d_ws;
  float* q    = ws;            float* k  = ws + 1024;
  float* v    = ws + 2048;     float* yh = ws + 3072;
  float* cs   = ws + 4096;
  int*   tidx = (int*)(ws + 4352);
  float* s    = ws + 8192;                         // 16 x 65536 f32
  int*   idxbuf = (int*)(ws + 8192 + NH * SPITCH); // 16 x 12288 int

  gemv1024<<<128, 256, 0, stream>>>(Wr, x, q);
  gemv1024<<<128, 256, 0, stream>>>(Wk, x, k);
  gemv1024<<<128, 256, 0, stream>>>(Wv, x, v);

  score_wmma<<<1024, 128, 4 * 65536 + 2048, stream>>>(kc, q, s);

  topk_head<<<16, 1024, 16384 * 8 + 1088, stream>>>(s, idxbuf);

  chunk_scores<<<NCH, 256, 0, stream>>>(s, idxbuf, cs);
  topk_chunks<<<1, 256, 0, stream>>>(cs, tidx);

  final_attn<<<16, 256, 0, stream>>>(s, idxbuf, tidx, q, k, v, vc, yh);

  gemv1024<<<128, 256, 0, stream>>>(Wo, yh, (float*)d_out);
}
